// L2Attention_33509334843619
// MI455X (gfx1250) — matrix operand order repaired and checked by reading, not compiled, not run on hardware
//
#include <hip/hip_runtime.h>
#include <hip/hip_bf16.h>
#include <math.h>

// Problem constants (B=1)
#define BN 2048   // sequence length (= B*S)
#define HH 768    // hidden
#define NH 12     // "heads" (flat groups of 2048 rows of the [24576][64] q array)
#define HD 64     // head dim

typedef __attribute__((ext_vector_type(16))) __bf16 v16bf;
typedef __attribute__((ext_vector_type(8)))  __bf16 v8bf;
typedef __attribute__((ext_vector_type(8)))  float  v8f;

// log2(e)
#define LOG2E 1.4426950408889634f

__device__ __forceinline__ float fast_exp2(float x) {
#if __has_builtin(__builtin_amdgcn_exp2f)
  return __builtin_amdgcn_exp2f(x);   // v_exp_f32 (TRANS, co-executes with WMMA)
#else
  return exp2f(x);
#endif
}

// ---------------------------------------------------------------------------
// WMMA fragment loaders (wave32); A and B share contiguous k order k0 + hi*16 + i.
//
// A (16x32 bf16, M rows x K): lane l (r=l&15, hi=l>>4) holds
//   elems 0..7  -> row r, K = k0 + hi*16 + i        (16B contiguous)
//   elems 8..15 -> row r, K = k0 + hi*16 + i        (16B contiguous)
__device__ __forceinline__ v16bf load_a_frag(const __bf16* p, int ld, int row0, int k0) {
  int lane = threadIdx.x & 31;
  int r = lane & 15, hi = lane >> 4;
  const __bf16* base = p + (size_t)(row0 + r) * ld + k0 + hi * 16;
  v8bf lo = *(const v8bf*)(base);
  v8bf hh = *(const v8bf*)(base + 8);
  v16bf out;
#pragma unroll
  for (int i = 0; i < 8; i++) { out[i] = lo[i]; out[i + 8] = hh[i]; }
  return out;
}

// B (32x16 bf16, K x N). Source row-major [n][k] ("transposed" operand):
//   lane l (n=l&15, hi=l>>4) holds B[k0 + hi*16 + i][col0+n] = src[col0+n][k0+hi*16+i]
//   -> one contiguous 32B load per lane.
__device__ __forceinline__ v16bf load_b_frag(const __bf16* p, int ld, int col0, int k0) {
  int lane = threadIdx.x & 31;
  int n = lane & 15, hi = lane >> 4;
  return *(const v16bf*)(p + (size_t)(col0 + n) * ld + k0 + hi * 16);
}

__device__ __forceinline__ v8f wmma_bf16(v16bf a, v16bf b, v8f c) {
  return __builtin_amdgcn_wmma_f32_16x16x32_bf16(
      /*neg_a=*/false, a, /*neg_b=*/false, b,
      /*c_mod=*/(short)0, c, /*reuse_a=*/false, /*reuse_b=*/false);
}

// ---------------------------------------------------------------------------
// 1) fp32 -> bf16 conversion (grid-stride)
__global__ void cvt_bf16(const float* __restrict__ in, __bf16* __restrict__ out, int n) {
  int i = blockIdx.x * blockDim.x + threadIdx.x;
  int stride = gridDim.x * blockDim.x;
  for (; i < n; i += stride) out[i] = (__bf16)in[i];
}

// 2) Projection: out[n][o] = sum_k X[n][k] * W[o][k] + bias[o]   (2048x768, K=768)
__global__ __launch_bounds__(128) void proj_kernel(
    const __bf16* __restrict__ X, const __bf16* __restrict__ W,
    const float* __restrict__ bias, float* __restrict__ out) {
  int wave = threadIdx.x >> 5;
  int lane = threadIdx.x & 31;
  int nl = lane & 15, hi = lane >> 4;
  int col0 = (blockIdx.x * 4 + wave) * 16;
  int row0 = blockIdx.y * 16;
  v8f c = {};
#pragma unroll 4
  for (int k = 0; k < HH; k += 32) {
    v16bf a = load_a_frag(X, HH, row0, k);
    v16bf b = load_b_frag(W, HH, col0, k);
    c = wmma_bf16(a, b, c);
  }
  float bb = bias[col0 + nl];
#pragma unroll
  for (int e = 0; e < 8; e++) {  // C/D: elem e -> row e+8*hi, col nl
    out[(size_t)(row0 + e + 8 * hi) * HH + col0 + nl] = c[e] + bb;
  }
}

// 3) Normalize q rows (flat [24576][64]), emit bf16 q and fp32 AA per row.
__global__ void norm_q(const float* __restrict__ qtmp, __bf16* __restrict__ qn,
                       float* __restrict__ aa) {
  int j = blockIdx.x * blockDim.x + threadIdx.x;
  if (j >= BN * NH) return;
  const float* row = qtmp + (size_t)j * HD;
  float vals[HD];
  float s = 0.f;
#pragma unroll
  for (int d = 0; d < HD; d++) { vals[d] = row[d]; s += vals[d] * vals[d]; }
  float inv = 1.f / fmaxf(sqrtf(s), 1e-12f);
  float s2 = 0.f;
  __bf16* qrow = qn + (size_t)j * HD;
#pragma unroll
  for (int d = 0; d < HD; d++) {
    float q = vals[d] * inv;
    s2 += q * q;
    qrow[d] = (__bf16)q;
  }
  aa[j] = s2;  // AA for this row (reference computes it post-normalize)
}

// 4) v transpose per group: vt[h][d][m] = vtmp[m][h*64+d]  (bf16)
__global__ void transp_v(const float* __restrict__ vtmp, __bf16* __restrict__ vt) {
  int idx = blockIdx.x * blockDim.x + threadIdx.x;
  if (idx >= NH * HD * BN) return;
  int m = idx & (BN - 1);
  int t = idx >> 11;      // / 2048
  int d = t & 63;
  int h = t >> 6;
  vt[idx] = (__bf16)vtmp[(size_t)m * HH + h * HD + d];
}

// 5) Fused L2-distance attention for group h.
//    sim = (2AB - AA_r - AA_c)/8 in [-0.5, ~0] because q is unit-norm
//    => fixed-base softmax (no online max, no accumulator rescale).
//    Each wave owns 32 query rows (two 16-row tiles) so every Q/V B-fragment
//    feeds two WMMAs (halves L2 traffic per FLOP). 16 WMMA per 32-column step.
__global__ __launch_bounds__(128) void flash_kernel(
    const __bf16* __restrict__ qn, const float* __restrict__ aa,
    const __bf16* __restrict__ vt, float* __restrict__ out) {
  __shared__ __bf16 pbuf[4][2][16 * 32];  // per-wave, per-row-tile 16x32 P stage

  int wave = threadIdx.x >> 5;
  int lane = threadIdx.x & 31;
  int nl = lane & 15, hi = lane >> 4;
  int h = blockIdx.y;
  int row0 = blockIdx.x * 128 + wave * 32;   // 32 rows per wave

  const __bf16* qh  = qn + (size_t)h * BN * HD;   // [2048][64]
  const float*  aah = aa + h * BN;
  const __bf16* vth = vt + (size_t)h * HD * BN;   // [64][2048]

  const float C1 = 0.125f * LOG2E;   // folds 1/sqrt(64) and ln->log2
  const float C2 = 0.25f * LOG2E;    // 2 * C1 (coefficient of AB)

  // Preload the wave's Q rows as A-fragments: [tile t][k-chunk]
  v16bf aq00 = load_a_frag(qh, HD, row0, 0);
  v16bf aq01 = load_a_frag(qh, HD, row0, 32);
  v16bf aq10 = load_a_frag(qh, HD, row0 + 16, 0);
  v16bf aq11 = load_a_frag(qh, HD, row0 + 16, 32);

  float aar2[2][8];   // AA_row * C1 (pre-scaled)
#pragma unroll
  for (int t = 0; t < 2; t++)
#pragma unroll
    for (int e = 0; e < 8; e++)
      aar2[t][e] = aah[row0 + t * 16 + e + 8 * hi] * C1;

  v8f acc[2][4] = {};
  float lr[2][8] = {};   // per-lane partial softmax denominators

  for (int j = 0; j < BN; j += 32) {
    if (j + 32 < BN) __builtin_prefetch(vth + (size_t)nl * BN + j + 32, 0, 0);

    // ---- S = Q K^T: 2 row tiles x 2 col tiles, K=64 -> 8 WMMAs ----
    v16bf bq0 = load_b_frag(qh, HD, j, 0);
    v16bf bq1 = load_b_frag(qh, HD, j, 32);
    v16bf bq2 = load_b_frag(qh, HD, j + 16, 0);
    v16bf bq3 = load_b_frag(qh, HD, j + 16, 32);
    v8f s00 = {}, s01 = {}, s10 = {}, s11 = {};
    s00 = wmma_bf16(aq00, bq0, s00); s00 = wmma_bf16(aq01, bq1, s00);
    s01 = wmma_bf16(aq00, bq2, s01); s01 = wmma_bf16(aq01, bq3, s01);
    s10 = wmma_bf16(aq10, bq0, s10); s10 = wmma_bf16(aq11, bq1, s10);
    s11 = wmma_bf16(aq10, bq2, s11); s11 = wmma_bf16(aq11, bq3, s11);

    float off0 = aah[j + nl] * C1;        // AA_col * C1 for col sub-tile 0
    float off1 = aah[j + 16 + nl] * C1;   // col sub-tile 1

    // ---- probabilities: p = 2^(C2*AB - C1*AA_r - C1*AA_c) = exp(sim) ----
#pragma unroll
    for (int t = 0; t < 2; t++) {
      const v8f& sa = (t == 0) ? s00 : s10;
      const v8f& sb = (t == 0) ? s01 : s11;
      __bf16* pb = pbuf[wave][t];
#pragma unroll
      for (int e = 0; e < 8; e++) {
        float p0 = fast_exp2(fmaf(C2, sa[e], -(aar2[t][e] + off0)));
        float p1 = fast_exp2(fmaf(C2, sb[e], -(aar2[t][e] + off1)));
        lr[t][e] += p0 + p1;
        int r = e + 8 * hi;               // C/D layout -> row-major LDS stage
        pb[r * 32 + nl]      = (__bf16)p0;
        pb[r * 32 + 16 + nl] = (__bf16)p1;
      }
    }

    // ---- O += P (16x32) x V (32x64): 8 WMMAs, V frags reused across tiles ----
    v16bf ap0 = load_a_frag(pbuf[wave][0], 32, 0, 0);  // in-wave DS ordering
    v16bf ap1 = load_a_frag(pbuf[wave][1], 32, 0, 0);
#pragma unroll
    for (int d = 0; d < 4; d++) {
      v16bf bv = load_b_frag(vth, BN, d * 16, j);
      acc[0][d] = wmma_bf16(ap0, bv, acc[0][d]);
      acc[1][d] = wmma_bf16(ap1, bv, acc[1][d]);
    }
  }

  // epilogue: reduce per-lane denominators across the 16-lane half, normalize,
  // store. out flat [h*2048 + row][64] == final [2048][768] buffer.
#pragma unroll
  for (int t = 0; t < 2; t++) {
#pragma unroll
    for (int e = 0; e < 8; e++) {
      float r = lr[t][e];
      r += __shfl_xor(r, 1);
      r += __shfl_xor(r, 2);
      r += __shfl_xor(r, 4);
      r += __shfl_xor(r, 8);
      float inv = 1.f / r;
      size_t base = ((size_t)h * BN + row0 + t * 16 + e + 8 * hi) * HD;
      out[base + nl]      = acc[t][0][e] * inv;
      out[base + 16 + nl] = acc[t][1][e] * inv;
      out[base + 32 + nl] = acc[t][2][e] * inv;
      out[base + 48 + nl] = acc[t][3][e] * inv;
    }
  }
}

// ---------------------------------------------------------------------------
extern "C" void kernel_launch(void* const* d_in, const int* in_sizes, int n_in,
                              void* d_out, int out_size, void* d_ws, size_t ws_size,
                              hipStream_t stream) {
  const float* hs = (const float*)d_in[0];
  // d_in[1] (g) and d_in[2] (attention_mask) are dead in the reference — never read.
  const float* Wq = (const float*)d_in[3];
  const float* bq = (const float*)d_in[4];
  const float* Wv = (const float*)d_in[5];
  const float* bv = (const float*)d_in[6];
  float* out = (float*)d_out;

  // Workspace carve (~24.6 MB total), 256B aligned chunks.
  char* ws = (char*)d_ws;
  auto carve = [&](size_t bytes) {
    char* p = ws;
    ws += (bytes + 255) & ~(size_t)255;
    return p;
  };
  float*  qtmp  = (float*)carve((size_t)BN * HH * 4);
  float*  vtmp  = (float*)carve((size_t)BN * HH * 4);
  __bf16* hs_bf = (__bf16*)carve((size_t)BN * HH * 2);
  __bf16* wq_bf = (__bf16*)carve((size_t)HH * HH * 2);
  __bf16* wv_bf = (__bf16*)carve((size_t)HH * HH * 2);
  __bf16* qn    = (__bf16*)carve((size_t)NH * BN * HD * 2);
  __bf16* vt    = (__bf16*)carve((size_t)NH * BN * HD * 2);
  float*  aa    = (float*)carve((size_t)NH * BN * 4);

  cvt_bf16<<<dim3(512), dim3(256), 0, stream>>>(hs, hs_bf, BN * HH);
  cvt_bf16<<<dim3(512), dim3(256), 0, stream>>>(Wq, wq_bf, HH * HH);
  cvt_bf16<<<dim3(512), dim3(256), 0, stream>>>(Wv, wv_bf, HH * HH);

  proj_kernel<<<dim3(HH / 64, BN / 16), dim3(128), 0, stream>>>(hs_bf, wq_bf, bq, qtmp);
  proj_kernel<<<dim3(HH / 64, BN / 16), dim3(128), 0, stream>>>(hs_bf, wv_bf, bv, vtmp);

  norm_q<<<dim3((BN * NH + 255) / 256), dim3(256), 0, stream>>>(qtmp, qn, aa);
  transp_v<<<dim3((NH * HD * BN + 255) / 256), dim3(256), 0, stream>>>(vtmp, vt);

  flash_kernel<<<dim3(BN / 128, NH), dim3(128), 0, stream>>>(qn, aa, vt, out);
}
